// EmbededNeRF_21242908246583
// MI455X (gfx1250) — matrix operand order repaired and checked by reading, not compiled, not run on hardware
//
#include <hip/hip_runtime.h>
#include <math.h>

// ---------------------------------------------------------------------------
// Fused NeRF MLP inference for gfx1250 (MI455X), wave32 + WMMA f16.
//
//   - 2048 workgroups x 256 threads (8 waves); 64 points per workgroup.
//   - Wave pairs: each wave owns 16 rows x HALF of the output N-tiles.
//   - Weight k-slices are software-pipelined THROUGH REGISTERS: while the
//     WMMAs of slice kk run, the global loads of slice kk+1 are in flight;
//     the only consumer is the ds_store at the top of the next iteration.
//   - ALL B fragments of a k-step are preloaded, then a sched_barrier(0)
//     pins them before the WMMA block: one clause of ds_load_b128, ONE
//     s_wait_dscnt, then NTW back-to-back independent v_wmma.
//   - Activations live entirely in LDS (64 rows x 328 halves).
// ---------------------------------------------------------------------------

#define N_PTS    131072
#define BLK_PTS  64          // points per workgroup
#define NTHREADS 256         // 8 waves (wave32)
#define AS       328         // activation LDS row stride, in halves (bank-skewed)

typedef _Float16 h16v __attribute__((ext_vector_type(16)));
typedef _Float16 h8v  __attribute__((ext_vector_type(8)));
typedef float    f8v  __attribute__((ext_vector_type(8)));

struct NerfParams {
    const float* x;
    const float* v;
    const float* pts_b[9];
    const float* sigma_W;
    const float* sigma_b;
    const float* rgb_b0;
    const float* rgb_b1;
    const float* rgb_W2;
    const float* rgb_b2;
    const _Float16* frag;    // f16 fragment-ordered weights in d_ws
    float* out_rgb;
    float* out_sigma;
};

__device__ __forceinline__ void sched_fence() {
#if defined(__HIP_DEVICE_COMPILE__) && __has_builtin(__builtin_amdgcn_sched_barrier)
    __builtin_amdgcn_sched_barrier(0);   // nothing may be reordered across this
#endif
}

__device__ __forceinline__ f8v wmma_f16(h16v a, h16v b, f8v c) {
#if defined(__HIP_DEVICE_COMPILE__)
    // D = A(16x32 f16) * B(32x16 f16) + C(16x16 f32)
    return __builtin_amdgcn_wmma_f32_16x16x32_f16(
        /*neg_a=*/false, a, /*neg_b=*/false, b,
        /*c_mod=*/(short)0, c, /*reuse_a=*/false, /*reuse_b=*/false);
#else
    return c;
#endif
}

// One dense layer: s_act[:, 0:32*ksteps] (f16) @ W -> s_act[:, 0:NTOT*16] (f16)
// NTOT = total 16-wide output tiles (16 -> fout=256, 8 -> fout=128).
// NTW  = NTOT/2 tiles handled by this wave (wave pairs split the N dimension).
// Weight fragments are pre-packed: slice kk is NTOT*512 contiguous halves,
// within a slice: nt*512 + lane*16 + e  (lane = (n&15)|(k&16), e = k&15).
template <int NTOT>
__device__ __forceinline__ void wmma_layer(const _Float16* __restrict__ wfrag,
                                           const float* __restrict__ bias,
                                           int ksteps, bool relu,
                                           _Float16* s_act, _Float16* s_w,
                                           int tid, int lane, int rowBase,
                                           int colTile)
{
    constexpr int NTW  = NTOT / 2;               // tiles per wave
    constexpr int ITER = (NTOT * 64) / NTHREADS; // h8v chunks per thread (4 / 2)

    f8v acc[NTW];
    const f8v zero = {0.f, 0.f, 0.f, 0.f, 0.f, 0.f, 0.f, 0.f};
#pragma unroll
    for (int nt = 0; nt < NTW; ++nt) acc[nt] = zero;

    // Prologue: fetch weight slice 0 into registers (in flight over phase-0).
    h8v stage[ITER];
    {
        const h8v* g = (const h8v*)wfrag;
#pragma unroll
        for (int i = 0; i < ITER; ++i) stage[i] = g[tid + i * NTHREADS];
    }

#pragma unroll 1
    for (int kk = 0; kk < ksteps; ++kk) {
        __syncthreads();                         // previous slice fully consumed
        {
            // Commit slice kk (registers -> LDS); this is the only point that
            // must wait on the global loads issued one iteration ago.
            h8v* l = (h8v*)s_w;
#pragma unroll
            for (int i = 0; i < ITER; ++i) l[tid + i * NTHREADS] = stage[i];
        }
        if (kk + 1 < ksteps) {
            // Launch slice kk+1 now; its latency hides under this slice's WMMAs.
            const h8v* g = (const h8v*)(wfrag + (size_t)(kk + 1) * NTOT * 512);
#pragma unroll
            for (int i = 0; i < ITER; ++i) stage[i] = g[tid + i * NTHREADS];
        }
        __syncthreads();                         // slice kk visible to all waves

        // A fragment (16x32 f16): lanes 0-15 = rows, hold K[0..15];
        // lanes 16-31 hold K[16..31].
        union { h16v v; h8v p[2]; } a;
        const _Float16* ap = s_act + (size_t)(rowBase + (lane & 15)) * AS
                                   + kk * 32 + ((lane >> 4) << 4);
        a.p[0] = *(const h8v*)ap;
        a.p[1] = *(const h8v*)(ap + 8);

        // Preload ALL B fragments for this k-step, then pin them before the
        // WMMA block with a scheduling fence: the loads form one clause with
        // a single DS wait and cannot be sunk/coalesced by the scheduler.
        union BU { h16v v; h8v p[2]; };
        BU b[NTW];
        const int bbase = colTile * 512 + lane * 16;
#pragma unroll
        for (int nt = 0; nt < NTW; ++nt) {
            const h8v* bp = (const h8v*)(s_w + bbase + nt * 512);
            b[nt].p[0] = bp[0];
            b[nt].p[1] = bp[1];
        }
        sched_fence();
        // Back-to-back independent WMMAs (independent accumulators).
#pragma unroll
        for (int nt = 0; nt < NTW; ++nt)
            acc[nt] = wmma_f16(a.v, b[nt].v, acc[nt]);
    }

    // Epilogue: bias + (optional) ReLU, f32 -> f16, back to this wave's rows
    // and its own column half (disjoint across the wave pair -> no hazards).
    // C/D layout: VGPR r, lanes 0-15 -> M=r, lanes 16-31 -> M=r+8; N = lane&15.
#pragma unroll
    for (int nt = 0; nt < NTW; ++nt) {
        const float bv = bias[(colTile + nt) * 16 + (lane & 15)];
#pragma unroll
        for (int r = 0; r < 8; ++r) {
            float val = acc[nt][r] + bv;
            if (relu) val = fmaxf(val, 0.f);
            s_act[(size_t)(rowBase + r + ((lane >> 4) << 3)) * AS
                  + (colTile + nt) * 16 + (lane & 15)] = (_Float16)val;
        }
    }
}

__launch_bounds__(NTHREADS, 1)
__global__ void nerf_fused_kernel(NerfParams p)
{
    __shared__ _Float16 s_w[16 * 512];          // 16 KB: one weight k-slice
    __shared__ _Float16 s_act[BLK_PTS * AS];    // ~41 KB: activations

    const int tid = threadIdx.x;
    const int lane = tid & 31;
    const int wave = tid >> 5;                  // 0..7
    const int rowBase = (wave >> 1) * 16;       // wave pair shares 16 rows
    const int nHalf = wave & 1;                 // which half of the N tiles
    const int gbase = blockIdx.x * BLK_PTS;

    // ---- Phase 0: positional encoding -> cols [0,64) and a copy at [256,320)
    // (the copy feeds the skip-concat layer; cols 256+ are untouched until then)
    if (tid < BLK_PTS) {
        const int gp = gbase + tid;
        float xv[3] = { p.x[3 * gp + 0], p.x[3 * gp + 1], p.x[3 * gp + 2] };
        float pe[64];
        pe[0] = xv[0]; pe[1] = xv[1]; pe[2] = xv[2];
        float fr = 1.f;
        for (int f = 0; f < 10; ++f) {
            for (int c = 0; c < 3; ++c) {
                const float ang = xv[c] * fr;
                pe[3 + f * 3 + c]  = __sinf(ang);
                pe[33 + f * 3 + c] = __cosf(ang);
            }
            fr *= 2.f;
        }
        pe[63] = 0.f;                            // K padding 63 -> 64
        for (int j = 0; j < 64; ++j) {
            const _Float16 hv = (_Float16)pe[j];
            s_act[tid * AS + j] = hv;
            s_act[tid * AS + 256 + j] = hv;
        }
    }

    // ---- pts MLP: 8 hidden (ReLU) + output linear (no activation)
    wmma_layer<16>(p.frag + 0,      p.pts_b[0], 2,  true,  s_act, s_w, tid, lane, rowBase, nHalf * 8);
    wmma_layer<16>(p.frag + 16384,  p.pts_b[1], 8,  true,  s_act, s_w, tid, lane, rowBase, nHalf * 8);
    wmma_layer<16>(p.frag + 81920,  p.pts_b[2], 8,  true,  s_act, s_w, tid, lane, rowBase, nHalf * 8);
    wmma_layer<16>(p.frag + 147456, p.pts_b[3], 8,  true,  s_act, s_w, tid, lane, rowBase, nHalf * 8);
    wmma_layer<16>(p.frag + 212992, p.pts_b[4], 10, true,  s_act, s_w, tid, lane, rowBase, nHalf * 8); // skip 319->320
    wmma_layer<16>(p.frag + 294912, p.pts_b[5], 8,  true,  s_act, s_w, tid, lane, rowBase, nHalf * 8);
    wmma_layer<16>(p.frag + 360448, p.pts_b[6], 8,  true,  s_act, s_w, tid, lane, rowBase, nHalf * 8);
    wmma_layer<16>(p.frag + 425984, p.pts_b[7], 8,  true,  s_act, s_w, tid, lane, rowBase, nHalf * 8);
    wmma_layer<16>(p.frag + 491520, p.pts_b[8], 8,  false, s_act, s_w, tid, lane, rowBase, nHalf * 8);

    __syncthreads();                             // h final, visible to all waves

    // ---- sigma head + SH4 view embedding into cols [256,288)
    if (tid < BLK_PTS) {
        const int gp = gbase + tid;
        float s = p.sigma_b[0];
        for (int c = 0; c < 256; ++c)
            s += (float)s_act[tid * AS + c] * p.sigma_W[c];
        p.out_sigma[gp] = s;

        const float X = p.v[3 * gp + 0], Y = p.v[3 * gp + 1], Z = p.v[3 * gp + 2];
        const float xx = X * X, yy = Y * Y, zz = Z * Z;
        float sh[16];
        sh[0]  = 0.28209479177387814f;
        sh[1]  = -0.48860251190291987f * Y;
        sh[2]  =  0.48860251190291987f * Z;
        sh[3]  = -0.48860251190291987f * X;
        sh[4]  =  1.0925484305920792f * X * Y;
        sh[5]  = -1.0925484305920792f * Y * Z;
        sh[6]  =  0.94617469575755997f * zz - 0.31539156525251999f;
        sh[7]  = -1.0925484305920792f * X * Z;
        sh[8]  =  0.54627421529603959f * (xx - yy);
        sh[9]  = -0.59004358992664352f * Y * (3.f * xx - yy);
        sh[10] =  2.8906114426405538f * X * Y * Z;
        sh[11] =  0.45704579946446572f * Y * (1.f - 5.f * zz);
        sh[12] =  0.3731763325901154f * Z * (5.f * zz - 3.f);
        sh[13] =  0.45704579946446572f * X * (1.f - 5.f * zz);
        sh[14] =  1.4453057213202769f * Z * (xx - yy);
        sh[15] = -0.59004358992664352f * X * (xx - 3.f * yy);
        for (int j = 0; j < 16; ++j)
            s_act[tid * AS + 256 + j] = (_Float16)sh[j];
        for (int j = 16; j < 32; ++j)                 // K padding 272 -> 288
            s_act[tid * AS + 256 + j] = (_Float16)0.f;
    }
    // (rgb0's internal barrier makes the SH writes visible before A-reads)

    // ---- rgb MLP: 272(->288) -> 128 -> 128 (WMMA), 128 -> 3 (VALU)
    wmma_layer<8>(p.frag + 557056, p.rgb_b0, 9, true, s_act, s_w, tid, lane, rowBase, nHalf * 4);
    wmma_layer<8>(p.frag + 593920, p.rgb_b1, 4, true, s_act, s_w, tid, lane, rowBase, nHalf * 4);

    __syncthreads();

    if (tid < BLK_PTS) {
        const int gp = gbase + tid;
        float o[3] = { p.rgb_b2[0], p.rgb_b2[1], p.rgb_b2[2] };
        for (int c = 0; c < 128; ++c) {
            const float hv = (float)s_act[tid * AS + c];
            o[0] += hv * p.rgb_W2[c * 3 + 0];
            o[1] += hv * p.rgb_W2[c * 3 + 1];
            o[2] += hv * p.rgb_W2[c * 3 + 2];
        }
        for (int j = 0; j < 3; ++j)
            p.out_rgb[gp * 3 + j] = 1.f / (1.f + __expf(-o[j]));
    }
}

// ---------------------------------------------------------------------------
// Kernel 1: convert f32 [fin x Nout] weights to f16 in WMMA B-fragment order,
// zero-padding K up to Kpad. Fragment element (k, n) lands at
//   slice(k/32)*ntiles*512 + (n/16)*512 + lane*16 + e,
// lane = (n&15) | (((k&31)>>4)<<4), e = k&15  -- contiguous k order, same as the A-fragment loads.
// ---------------------------------------------------------------------------
__global__ void convert_weights_kernel(const float* __restrict__ src,
                                       _Float16* __restrict__ dst,
                                       int fin, int Nout, int Kpad)
{
    const int idx = blockIdx.x * blockDim.x + threadIdx.x;
    const int total = Kpad * Nout;
    if (idx >= total) return;
    const int k = idx / Nout;
    const int n = idx - k * Nout;
    const float w = (k < fin) ? src[k * Nout + n] : 0.f;
    const int kk = k >> 5, kin = k & 31;
    const int nt = n >> 4, nn = n & 15;
    const int lane = nn | ((kin >> 4) << 4);
    const int e = kin & 15;
    const int ntiles = Nout >> 4;
    const size_t off = ((size_t)kk * ntiles + nt) * 512 + (size_t)lane * 16 + e;
    dst[off] = (_Float16)w;
}

extern "C" void kernel_launch(void* const* d_in, const int* in_sizes, int n_in,
                              void* d_out, int out_size, void* d_ws, size_t ws_size,
                              hipStream_t stream)
{
    (void)in_sizes; (void)n_in; (void)out_size; (void)ws_size;

    // setup_inputs() order: x, v, pts_Ws[0..8], pts_bs[0..8],
    //                       sigma_W, sigma_b, rgb_Ws[0..2], rgb_bs[0..2]
    const float* x = (const float*)d_in[0];
    const float* v = (const float*)d_in[1];
    const float* ptsW[9];
    const float* ptsB[9];
    for (int i = 0; i < 9; ++i) {
        ptsW[i] = (const float*)d_in[2 + i];
        ptsB[i] = (const float*)d_in[11 + i];
    }
    const float* sigW  = (const float*)d_in[20];
    const float* sigB  = (const float*)d_in[21];
    const float* rgbW0 = (const float*)d_in[22];
    const float* rgbW1 = (const float*)d_in[23];
    const float* rgbW2 = (const float*)d_in[24];
    const float* rgbB0 = (const float*)d_in[25];
    const float* rgbB1 = (const float*)d_in[26];
    const float* rgbB2 = (const float*)d_in[27];

    _Float16* frag = (_Float16*)d_ws;           // ~1.22 MB of f16 fragments

    struct Cvt { const float* src; int fin, Nout, Kpad; size_t off; };
    const Cvt cv[11] = {
        { ptsW[0],  63, 256,  64, 0      },
        { ptsW[1], 256, 256, 256, 16384  },
        { ptsW[2], 256, 256, 256, 81920  },
        { ptsW[3], 256, 256, 256, 147456 },
        { ptsW[4], 319, 256, 320, 212992 },     // skip layer (h ++ pe)
        { ptsW[5], 256, 256, 256, 294912 },
        { ptsW[6], 256, 256, 256, 360448 },
        { ptsW[7], 256, 256, 256, 425984 },
        { ptsW[8], 256, 256, 256, 491520 },     // output linear
        { rgbW0,   272, 128, 288, 557056 },     // rgb0 (h ++ sh4)
        { rgbW1,   128, 128, 128, 593920 },     // rgb1
    };
    for (int i = 0; i < 11; ++i) {
        const int total = cv[i].Kpad * cv[i].Nout;
        const int grid = (total + 255) / 256;
        convert_weights_kernel<<<grid, 256, 0, stream>>>(
            cv[i].src, frag + cv[i].off, cv[i].fin, cv[i].Nout, cv[i].Kpad);
    }

    NerfParams p;
    p.x = x; p.v = v;
    for (int i = 0; i < 9; ++i) p.pts_b[i] = ptsB[i];
    p.sigma_W = sigW; p.sigma_b = sigB;
    p.rgb_b0 = rgbB0; p.rgb_b1 = rgbB1;
    p.rgb_W2 = rgbW2; p.rgb_b2 = rgbB2;
    p.frag = frag;
    p.out_rgb = (float*)d_out;
    p.out_sigma = (float*)d_out + (size_t)N_PTS * 3;

    nerf_fused_kernel<<<N_PTS / BLK_PTS, NTHREADS, 0, stream>>>(p);
}
